// QuantLinear_82463372083492
// MI455X (gfx1250) — matrix operand order repaired and checked by reading, not compiled, not run on hardware
//
#include <hip/hip_runtime.h>

// ---------------------------------------------------------------------------
// GPTQ W4A32 linear for MI455X (gfx1250, wave32).
//   out[128,11008] = x[128,4096] @ dequant(qweight,qzeros,scales) + bias
//
// prep1: x f32 -> f16 table in ws (1 MB; x lives in L2 and is reused by all
//        172 blocks, so activation conversion leaves the hot loop).
// prep2: per-(g,n) coeff {s, c=-(1024+z')*s} packed as 2xf16 in u32 (1.4 MB).
// main : 128(M) x 64(N) block tile, BK=32, double-buffered LDS, 8 wave32
//        waves in 4(M) x 2(N); each wave owns 32x32 = 2x2 subtiles ->
//        4 v_wmma_f32_16x16x32_f16 per step from 2 A-frags + 2 B-frags.
//        Dequant: half(1024+q) = 0x6400|nibble, w = fma(hq, s, c) via
//        v_pk_fma_f16 -> no int->float converts in the hot loop.
//        172 blocks keep all WGPs busy; 32 acc VGPRs keep occupancy high.
// ---------------------------------------------------------------------------

typedef _Float16 v16h __attribute__((ext_vector_type(16)));
typedef _Float16 v8h  __attribute__((ext_vector_type(8)));
typedef _Float16 v4h  __attribute__((ext_vector_type(4)));
typedef _Float16 v2h  __attribute__((ext_vector_type(2)));
typedef float    v8f  __attribute__((ext_vector_type(8)));

#define M_   128
#define K_   4096
#define N_   11008
#define PACK 8
#define G_   32          // K / group_size
#define BM   128
#define BN   64
#define BK   32
#define NSTEP (K_ / BK)  // 128

#define ASTRIDE 40                        // halves/row: 32 + 8 pad (80B, conflict-free b128)
#define BSTRIDE 40
#define A_HALVES (BM * ASTRIDE)           // 5120
#define B_HALVES (BN * BSTRIDE)           // 2560
#define BUF_HALVES (A_HALVES + B_HALVES)  // 7680 halves = 15360 B / buffer

// ---------------- prep kernel 1: x f32 -> f16 ----------------
__global__ void __launch_bounds__(256)
cvt_x_kernel(const float* __restrict__ x, _Float16* __restrict__ x16)
{
    const int i4 = blockIdx.x * 256 + threadIdx.x;   // float4 index, 131072 total
    const float4 v = *(const float4*)(x + i4 * 4);
    v4h h = { (_Float16)v.x, (_Float16)v.y, (_Float16)v.z, (_Float16)v.w };
    *(v4h*)(x16 + i4 * 4) = h;
}

// ---------------- prep kernel 2: dequant coefficients ----------------
// coeff[g*N+n] = pack( half(s), half(-(1024 + z') * s) ), z' = unpacked_zero + 1
__global__ void __launch_bounds__(256)
coeff_kernel(const unsigned int* __restrict__ qzeros,
             const float* __restrict__ scales,
             unsigned int* __restrict__ coeff)
{
    const int n = blockIdx.x * 256 + threadIdx.x;    // 0..11007 (43*256)
    const int g = blockIdx.y;                        // 0..31
    const unsigned int z =
        ((qzeros[g * (N_ / PACK) + (n >> 3)] >> ((n & 7) * 4)) & 15u) + 1u;
    const float s = scales[g * N_ + n];
    const float c = -(1024.0f + (float)z) * s;
    union { _Float16 h; unsigned short u; } hs, hc;
    hs.h = (_Float16)s;
    hc.h = (_Float16)c;
    coeff[g * N_ + n] = (unsigned int)hs.u | ((unsigned int)hc.u << 16);
}

// ---------------- helpers ----------------
__device__ inline v2h bits2h(unsigned int u) {
    union { unsigned int u; v2h h; } t; t.u = u; return t.h;
}

// 8 nibbles of qv -> 8 f16 weights via (0x6400|q) * s + c  (4x v_pk_fma_f16)
__device__ inline v8h dequant8(unsigned int qv, unsigned int cf) {
    const v2h ps = bits2h((cf & 0xFFFFu) * 0x10001u);   // (s, s)
    const v2h pc = bits2h((cf >> 16) * 0x10001u);       // (c, c)
    union { v8h v; v2h p[4]; } r;
#pragma unroll
    for (int p = 0; p < 4; ++p) {
        const unsigned int v = qv >> (8 * p);
        const unsigned int t =
            (v & 0xFu) | ((v << 12) & 0xF0000u) | 0x64006400u;
        r.p[p] = bits2h(t) * ps + pc;                   // v_pk_fma_f16
    }
    return r.v;
}

// ---------------- main GEMM kernel ----------------
__global__ void __launch_bounds__(256)
gptq_wmma_kernel(const _Float16* __restrict__ x16,
                 const unsigned int* __restrict__ qweight,
                 const unsigned int* __restrict__ coeff,
                 const float* __restrict__ bias,
                 float* __restrict__ out)
{
    __shared__ _Float16 lds[2 * BUF_HALVES];   // 30720 B of 320 KB WGP LDS

    const int tid    = threadIdx.x;
    const int wave   = tid >> 5;
    const int lane   = tid & 31;
    const int laneM  = lane & 15;
    const int hi     = lane >> 4;          // WMMA fragment lane-half
    const int wave_m = (wave & 3) * 32;    // 4 waves across M
    const int wave_n = (wave >> 2) * 32;   // 2 waves across N
    const int nb     = blockIdx.x * BN;

    // A staging: 128x32 halves = 512 v8h units; 2 per thread
    int ar[2], ak[2];
#pragma unroll
    for (int i = 0; i < 2; ++i) {
        const int u = tid + 256 * i;
        ar[i] = u >> 2;          // row 0..127
        ak[i] = (u & 3) * 8;     // k offset 0,8,16,24
    }
    // B staging: 4x64 int32 = 256; 1 per thread
    const int qr  = tid >> 6;    // qweight row within step (k = qr*8 + j)
    const int ncl = tid & 63;    // column within N tile

    // ---- prefetch + stage step 0 ----
    v8h av[2];
    unsigned int qv, cfp;
    {
#pragma unroll
        for (int i = 0; i < 2; ++i)
            av[i] = *(const v8h*)(x16 + ar[i] * K_ + ak[i]);
        qv  = qweight[qr * N_ + nb + ncl];
        cfp = coeff[nb + ncl];             // group 0 coefficients (persist 4 steps)
    }
    {
        _Float16* Aw = lds;
        _Float16* Bw = lds + A_HALVES;
#pragma unroll
        for (int i = 0; i < 2; ++i)
            *(v8h*)(Aw + ar[i] * ASTRIDE + ak[i]) = av[i];
        *(v8h*)(Bw + ncl * BSTRIDE + qr * 8) = dequant8(qv, cfp);
    }
    __syncthreads();

    v8f acc[2][2] = {};   // 32 accumulator VGPRs

    // ---------------- main K loop, double-buffered ----------------
    for (int s = 0; s < NSTEP; ++s) {
        const int buf = s & 1;

        // global prefetch for step s+1 (overlaps WMMA burst)
        v8h nav[2];
        unsigned int nqv = 0;
        if (s + 1 < NSTEP) {
            const int kb = (s + 1) * BK;
#pragma unroll
            for (int i = 0; i < 2; ++i)
                nav[i] = *(const v8h*)(x16 + ar[i] * K_ + kb + ak[i]);
            nqv = qweight[(kb / PACK + qr) * N_ + nb + ncl];
            if (((s + 1) & 3) == 0)        // group boundary every 4 steps
                cfp = coeff[((s + 1) >> 2) * N_ + nb + ncl];
        }

        // ---- compute: gather ALL fragments, then 4 back-to-back WMMAs ----
        {
            const _Float16* As = lds + buf * BUF_HALVES;
            const _Float16* Bs = As + A_HALVES;

            v16h afrag[2], bfrag[2];
#pragma unroll
            for (int mt = 0; mt < 2; ++mt) {
                const int aoff = (wave_m + mt * 16 + laneM) * ASTRIDE + hi * 8;
                const v8h alo = *(const v8h*)(As + aoff);
                const v8h ahi = *(const v8h*)(As + aoff + 16);
                afrag[mt] = __builtin_shufflevector(alo, ahi,
                    0, 1, 2, 3, 4, 5, 6, 7, 8, 9, 10, 11, 12, 13, 14, 15);
            }
#pragma unroll
            for (int nt = 0; nt < 2; ++nt) {
                const int boff = (wave_n + nt * 16 + laneM) * BSTRIDE + hi * 8;
                const v8h blo = *(const v8h*)(Bs + boff);
                const v8h bhi = *(const v8h*)(Bs + boff + 16);
                bfrag[nt] = __builtin_shufflevector(blo, bhi,
                    0, 1, 2, 3, 4, 5, 6, 7, 8, 9, 10, 11, 12, 13, 14, 15);
            }
#pragma unroll
            for (int mt = 0; mt < 2; ++mt)
#pragma unroll
                for (int nt = 0; nt < 2; ++nt)
                    acc[mt][nt] = __builtin_amdgcn_wmma_f32_16x16x32_f16(
                        false, afrag[mt], false, bfrag[nt],
                        (short)0, acc[mt][nt], false, false);
        }

        // ---- stage step s+1 into the other buffer ----
        if (s + 1 < NSTEP) {
            _Float16* Aw = lds + (buf ^ 1) * BUF_HALVES;
            _Float16* Bw = Aw + A_HALVES;
#pragma unroll
            for (int i = 0; i < 2; ++i)
                *(v8h*)(Aw + ar[i] * ASTRIDE + ak[i]) = nav[i];
            *(v8h*)(Bw + ncl * BSTRIDE + qr * 8) = dequant8(nqv, cfp);
            __syncthreads();
        }
    }

    // ---------------- epilogue: bias + f32 store ----------------
    // C/D layout: lanes 0-15 -> N=lane, M=v; lanes 16-31 -> N=lane-16, M=v+8
#pragma unroll
    for (int mt = 0; mt < 2; ++mt) {
#pragma unroll
        for (int nt = 0; nt < 2; ++nt) {
            const int n = nb + wave_n + nt * 16 + laneM;
            const float bv = bias[n];
#pragma unroll
            for (int v = 0; v < 8; ++v) {
                const int m = wave_m + mt * 16 + hi * 8 + v;
                out[m * N_ + n] = acc[mt][nt][v] + bv;
            }
        }
    }
}

extern "C" void kernel_launch(void* const* d_in, const int* in_sizes, int n_in,
                              void* d_out, int out_size, void* d_ws, size_t ws_size,
                              hipStream_t stream)
{
    const float*        xp = (const float*)d_in[0];
    const unsigned int* qw = (const unsigned int*)d_in[1];
    const unsigned int* qz = (const unsigned int*)d_in[2];
    const float*        sc = (const float*)d_in[3];
    const float*        bp = (const float*)d_in[4];
    // d_in[5] = g_idx: k // 128 by construction, folded into group = s>>2.
    float* outp = (float*)d_out;

    _Float16*     x16 = (_Float16*)d_ws;                          // 1 MB
    unsigned int* cft = (unsigned int*)((char*)d_ws + (size_t)M_ * K_ * 2);

    cvt_x_kernel<<<dim3((M_ * K_) / 4 / 256), dim3(256), 0, stream>>>(xp, x16);
    coeff_kernel<<<dim3(N_ / 256, G_), dim3(256), 0, stream>>>(qz, sc, cft);
    gptq_wmma_kernel<<<dim3(N_ / BN), dim3(256), 0, stream>>>(x16, qw, cft, bp, outp);
}
